// DCNv2SkipRefiner_52621939311392
// MI455X (gfx1250) — compile-verified, matrix operand order checked by reading, not run on hardware
//
#include <hip/hip_runtime.h>
#include <hip/hip_bf16.h>
#include <math.h>

// ---------------------------------------------------------------------------
// DCNv2 + InstanceNorm + ReLU for MI455X (gfx1250, wave32, WMMA).
// Both convolutions run as implicit GEMM on v_wmma_f32_16x16x32_bf16
// (bf16 A/B, f32 accumulate). K dimension ordered k-major (K = k*64 + c) so
// column-matrix builders emit packed ds_store_b128. Deformable gather runs on
// an NHWC copy of x (L2-resident) so each bilinear tap is 4x b128 loads.
// ---------------------------------------------------------------------------

typedef __bf16 bf16_t;
typedef bf16_t v8bf  __attribute__((ext_vector_type(8)));
typedef bf16_t v16bf __attribute__((ext_vector_type(16)));
typedef float  v8f   __attribute__((ext_vector_type(8)));

#define HH   128
#define WW   128
#define HWs  (128 * 128)
#define CIN  64
#define COUT 64
#define KDIM 576          // 9 taps * 64 channels  (K = k*64 + c)
#define KSTEPS 18         // 576 / 32
#define STR  584          // LDS row stride in bf16 elems (pad: conflict-free b128)

__device__ __forceinline__ bf16_t f2bf(float f) {
  union { float f; unsigned u; } v; v.f = f;
  unsigned r = v.u + 0x7FFFu + ((v.u >> 16) & 1u);   // round-to-nearest-even
  union { unsigned short s; bf16_t b; } o; o.s = (unsigned short)(r >> 16);
  return o.b;
}

union FragAB { v16bf v; v8bf h[2]; };

// 16-bit A/B matrix fragment per ISA layout:
// lanes 0-15 : row, K = kb+0..7 (VGPR0-3) and kb+16..23 (VGPR4-7)
// lanes16-31 : row, K = kb+8..15          and kb+24..31
__device__ __forceinline__ v16bf load_frag(const bf16_t* base, int row, int kb, int half) {
  FragAB f;
  const bf16_t* p = base + row * STR + kb + half * 8;
  f.h[0] = *(const v8bf*)(p);        // ds_load_b128
  f.h[1] = *(const v8bf*)(p + 16);   // ds_load_b128
  return f.v;
}

// ---------------------------------------------------------------------------
// Kernel 0: repack weights fp32 -> bf16, permuting K: wb[o][k*64+c]=w[o][c][k].
// ---------------------------------------------------------------------------
extern "C" __global__ void prep_kernel(const float* __restrict__ w_off,
                                       const float* __restrict__ w_dcn,
                                       bf16_t* __restrict__ wboff,
                                       bf16_t* __restrict__ wbdcn) {
  int i = blockIdx.x * 256 + threadIdx.x;
  if (i < 64 * KDIM) {
    int o = i / KDIM, r = i - o * KDIM;
    int k = r >> 6, c = r & 63;
    wbdcn[i] = f2bf(w_dcn[o * KDIM + c * 9 + k]);
  }
  if (i < 32 * KDIM) {
    int o = i / KDIM, r = i - o * KDIM;
    int k = r >> 6, c = r & 63;
    wboff[i] = (o < 27) ? f2bf(w_off[o * KDIM + c * 9 + k]) : f2bf(0.f);
  }
}

// ---------------------------------------------------------------------------
// Kernel 0b: NCHW -> NHWC transpose of x via LDS tile (coalesced both sides).
// ---------------------------------------------------------------------------
extern "C" __global__ __launch_bounds__(256)
void transpose_kernel(const float* __restrict__ x, float* __restrict__ xT) {
  __shared__ float tile[64][65];
  const int b    = blockIdx.y;
  const int pix0 = blockIdx.x * 64;
  const int t    = threadIdx.x;
  const float* xb = x + (size_t)b * CIN * HWs;
  for (int i = t; i < 64 * 64; i += 256) {
    int c = i >> 6, pc = i & 63;
    tile[c][pc] = xb[c * HWs + pix0 + pc];
  }
  __syncthreads();
  float* xo = xT + ((size_t)b * HWs + pix0) * CIN;
  for (int i = t; i < 64 * 64; i += 256) {
    int p = i >> 6, c = i & 63;
    xo[p * CIN + c] = tile[c][p];
  }
}

// ---------------------------------------------------------------------------
// Kernel 1: offset conv (27ch, padded to 32) via WMMA implicit GEMM.
// Tile: 4 rows x 16 cols of pixels (N=64), M=32 out channels, K=576.
// im2col builder packs 8 consecutive channels -> one ds_store_b128.
// ---------------------------------------------------------------------------
extern "C" __global__ __launch_bounds__(256)
void offset_conv_kernel(const float* __restrict__ x,
                        const bf16_t* __restrict__ wboff,
                        const float* __restrict__ b_off,
                        float* __restrict__ offs, float* __restrict__ mask) {
  extern __shared__ __align__(16) unsigned char smem[];
  bf16_t* Al = (bf16_t*)smem;        // 32 x STR  weights
  bf16_t* Bl = Al + 32 * STR;        // 64 x STR  im2col (p-major, K inner)

  const int b  = blockIdx.z;
  const int y0 = blockIdx.y * 4;
  const int x0 = blockIdx.x * 16;
  const int t  = threadIdx.x;

  {
    const unsigned* src = (const unsigned*)wboff;
    unsigned* dst = (unsigned*)Al;
    for (int i = t; i < 32 * 288; i += 256) {
      int r = i / 288, d = i - r * 288;
      dst[r * (STR / 2) + d] = src[r * 288 + d];
    }
  }
  const float* xb = x + (size_t)b * CIN * HWs;
  for (int i = t; i < 72 * 64; i += 256) {       // 9 taps * 8 c-groups * 64 px
    int p = i & 63, g = i >> 6;
    int k = g >> 3, cg = (g & 7) * 8;
    int ky = k / 3 - 1, kx = k - (k / 3) * 3 - 1;
    int gy = y0 + (p >> 4) + ky, gx = x0 + (p & 15) + kx;
    bool ok = (gy >= 0) && (gy < HH) && (gx >= 0) && (gx < WW);
    const float* src = xb + (cg * HWs + gy * WW + gx);
    v8bf pk;
    #pragma unroll
    for (int cc = 0; cc < 8; cc++) pk[cc] = f2bf(ok ? src[cc * HWs] : 0.f);
    *(v8bf*)(Bl + p * STR + k * 64 + cg) = pk;
  }
  __syncthreads();

  const int wave = t >> 5, lane = t & 31, half = lane >> 4, lr = lane & 15;
  const int m = wave & 1, n = wave >> 1;   // 2 M-tiles x 4 N-tiles = 8 waves
  v8f acc = {};
  for (int ks = 0; ks < KSTEPS; ks++) {
    int kb = ks * 32;
    v16bf a  = load_frag(Al, m * 16 + lr, kb, half);
    v16bf bb = load_frag(Bl, n * 16 + lr, kb, half);
    acc = __builtin_amdgcn_wmma_f32_16x16x32_bf16(false, a, false, bb,
                                                  (short)0, acc, false, false);
  }
  const int gpix = (y0 + n) * WW + (x0 + lr);    // p = n*16 + lr
  #pragma unroll
  for (int r = 0; r < 8; r++) {
    int oc = m * 16 + r + half * 8;
    if (oc < 18) {
      offs[((size_t)b * 18 + oc) * HWs + gpix] = acc[r] + b_off[oc];
    } else if (oc < 27) {
      float v = acc[r] + b_off[oc];
      mask[((size_t)b * 9 + (oc - 18)) * HWs + gpix] = 1.f / (1.f + expf(-v));
    }
  }
}

// ---------------------------------------------------------------------------
// Shared deform-conv WMMA tile loop + bias + store (layout-agnostic in K).
// ---------------------------------------------------------------------------
__device__ __forceinline__ void deform_mma_and_store(
    const bf16_t* Al, const bf16_t* Bl, const float* b_dcn, float* out,
    int b, int y0, int x0, int t) {
  const int wave = t >> 5, lane = t & 31, half = lane >> 4, lr = lane & 15;
  const int m = wave & 3, n0 = (wave >> 2) * 2, n1 = n0 + 1;
  v8f acc0 = {}, acc1 = {};
  for (int ks = 0; ks < KSTEPS; ks++) {
    int kb = ks * 32;
    v16bf a  = load_frag(Al, m * 16 + lr, kb, half);
    v16bf b0 = load_frag(Bl, n0 * 16 + lr, kb, half);
    v16bf b1 = load_frag(Bl, n1 * 16 + lr, kb, half);
    acc0 = __builtin_amdgcn_wmma_f32_16x16x32_bf16(false, a, false, b0,
                                                   (short)0, acc0, false, false);
    acc1 = __builtin_amdgcn_wmma_f32_16x16x32_bf16(false, a, false, b1,
                                                   (short)0, acc1, false, false);
  }
  #pragma unroll
  for (int r = 0; r < 8; r++) {
    int oc = m * 16 + r + half * 8;
    float bias = b_dcn[oc];
    int g0 = (y0 + n0) * WW + (x0 + lr);
    int g1 = (y0 + n1) * WW + (x0 + lr);
    out[((size_t)b * COUT + oc) * HWs + g0] = acc0[r] + bias;
    out[((size_t)b * COUT + oc) * HWs + g1] = acc1[r] + bias;
  }
}

// Per-(k,p) bilinear tap weights/indices (validity and mask folded in).
struct Taps { float A00, A01, A10, A11; int i00, i01, i10, i11; };
__device__ __forceinline__ Taps make_taps(float ys, float xs, float mk) {
  Taps tp;
  float yf = floorf(ys), xf = floorf(xs);
  int iy0 = (int)yf, ix0 = (int)xf, iy1 = iy0 + 1, ix1 = ix0 + 1;
  float wy1 = ys - yf, wy0 = 1.f - wy1;
  float wx1 = xs - xf, wx0 = 1.f - wx1;
  bool vy0 = (iy0 >= 0) & (iy0 < HH), vy1 = (iy1 >= 0) & (iy1 < HH);
  bool vx0 = (ix0 >= 0) & (ix0 < WW), vx1 = (ix1 >= 0) & (ix1 < WW);
  int cy0 = min(max(iy0, 0), HH - 1), cy1 = min(max(iy1, 0), HH - 1);
  int cx0 = min(max(ix0, 0), WW - 1), cx1 = min(max(ix1, 0), WW - 1);
  tp.A00 = mk * wy0 * wx0 * ((vy0 & vx0) ? 1.f : 0.f);
  tp.A01 = mk * wy0 * wx1 * ((vy0 & vx1) ? 1.f : 0.f);
  tp.A10 = mk * wy1 * wx0 * ((vy1 & vx0) ? 1.f : 0.f);
  tp.A11 = mk * wy1 * wx1 * ((vy1 & vx1) ? 1.f : 0.f);
  tp.i00 = cy0 * WW + cx0; tp.i01 = cy0 * WW + cx1;
  tp.i10 = cy1 * WW + cx0; tp.i11 = cy1 * WW + cx1;
  return tp;
}

// ---------------------------------------------------------------------------
// Kernel 2a: deformable conv, NHWC gather path (b128 tap loads, b128 stores).
// ---------------------------------------------------------------------------
extern "C" __global__ __launch_bounds__(256)
void deform_conv_nhwc_kernel(const float* __restrict__ xT,
                             const bf16_t* __restrict__ wbdcn,
                             const float* __restrict__ b_dcn,
                             const float* __restrict__ offs,
                             const float* __restrict__ mask,
                             float* __restrict__ out) {
  extern __shared__ __align__(16) unsigned char smem[];
  bf16_t* Al = (bf16_t*)smem;        // 64 x STR  weights
  bf16_t* Bl = Al + 64 * STR;        // 64 x STR  sampled columns

  const int b  = blockIdx.z;
  const int y0 = blockIdx.y * 4;
  const int x0 = blockIdx.x * 16;
  const int t  = threadIdx.x;

  {
    const unsigned* src = (const unsigned*)wbdcn;
    unsigned* dst = (unsigned*)Al;
    for (int i = t; i < 64 * 288; i += 256) {
      int r = i / 288, d = i - r * 288;
      dst[r * (STR / 2) + d] = src[r * 288 + d];
    }
  }

  const float* xTb = xT   + (size_t)b * HWs * CIN;   // [pix][c]
  const float* ofb = offs + (size_t)b * 18  * HWs;
  const float* mkb = mask + (size_t)b * 9   * HWs;
  const int p  = t >> 2;
  const int c0 = (t & 3) * 16;
  const int gy = y0 + (p >> 4), gx = x0 + (p & 15);
  const int pix = gy * WW + gx;
  #pragma unroll
  for (int k = 0; k < 9; k++) {
    const int ky = k / 3 - 1, kx = k - (k / 3) * 3 - 1;
    float oy = ofb[(2 * k) * HWs + pix];
    float ox = ofb[(2 * k + 1) * HWs + pix];
    float mk = mkb[k * HWs + pix];
    Taps tp = make_taps((float)(gy + ky) + oy, (float)(gx + kx) + ox, mk);
    const float4* p00 = (const float4*)(xTb + ((size_t)tp.i00 << 6) + c0);
    const float4* p01 = (const float4*)(xTb + ((size_t)tp.i01 << 6) + c0);
    const float4* p10 = (const float4*)(xTb + ((size_t)tp.i10 << 6) + c0);
    const float4* p11 = (const float4*)(xTb + ((size_t)tp.i11 << 6) + c0);
    v8bf lo, hi;
    #pragma unroll
    for (int j = 0; j < 4; j++) {
      float4 a = p00[j], b4 = p01[j], c4 = p10[j], d4 = p11[j];
      bf16_t e0 = f2bf(tp.A00 * a.x + tp.A01 * b4.x + tp.A10 * c4.x + tp.A11 * d4.x);
      bf16_t e1 = f2bf(tp.A00 * a.y + tp.A01 * b4.y + tp.A10 * c4.y + tp.A11 * d4.y);
      bf16_t e2 = f2bf(tp.A00 * a.z + tp.A01 * b4.z + tp.A10 * c4.z + tp.A11 * d4.z);
      bf16_t e3 = f2bf(tp.A00 * a.w + tp.A01 * b4.w + tp.A10 * c4.w + tp.A11 * d4.w);
      if (j < 2) { lo[j*4] = e0; lo[j*4+1] = e1; lo[j*4+2] = e2; lo[j*4+3] = e3; }
      else       { int q = (j-2)*4; hi[q] = e0; hi[q+1] = e1; hi[q+2] = e2; hi[q+3] = e3; }
    }
    bf16_t* dst = Bl + p * STR + k * 64 + c0;     // K = k*64 + c  (k-major)
    *(v8bf*)(dst)     = lo;                        // ds_store_b128
    *(v8bf*)(dst + 8) = hi;                        // ds_store_b128
  }
  __syncthreads();
  deform_mma_and_store(Al, Bl, b_dcn, out, b, y0, x0, t);
}

// ---------------------------------------------------------------------------
// Kernel 2b: deformable conv, NCHW gather fallback (if ws too small for xT).
// ---------------------------------------------------------------------------
extern "C" __global__ __launch_bounds__(256)
void deform_conv_kernel(const float* __restrict__ x,
                        const bf16_t* __restrict__ wbdcn,
                        const float* __restrict__ b_dcn,
                        const float* __restrict__ offs,
                        const float* __restrict__ mask,
                        float* __restrict__ out) {
  extern __shared__ __align__(16) unsigned char smem[];
  bf16_t* Al = (bf16_t*)smem;
  bf16_t* Bl = Al + 64 * STR;

  const int b  = blockIdx.z;
  const int y0 = blockIdx.y * 4;
  const int x0 = blockIdx.x * 16;
  const int t  = threadIdx.x;

  {
    const unsigned* src = (const unsigned*)wbdcn;
    unsigned* dst = (unsigned*)Al;
    for (int i = t; i < 64 * 288; i += 256) {
      int r = i / 288, d = i - r * 288;
      dst[r * (STR / 2) + d] = src[r * 288 + d];
    }
  }

  const float* xb  = x    + (size_t)b * CIN * HWs;
  const float* ofb = offs + (size_t)b * 18  * HWs;
  const float* mkb = mask + (size_t)b * 9   * HWs;
  const int p  = t >> 2;
  const int c0 = (t & 3) * 16;
  const int gy = y0 + (p >> 4), gx = x0 + (p & 15);
  const int pix = gy * WW + gx;
  #pragma unroll
  for (int k = 0; k < 9; k++) {
    const int ky = k / 3 - 1, kx = k - (k / 3) * 3 - 1;
    float oy = ofb[(2 * k) * HWs + pix];
    float ox = ofb[(2 * k + 1) * HWs + pix];
    float mk = mkb[k * HWs + pix];
    Taps tp = make_taps((float)(gy + ky) + oy, (float)(gx + kx) + ox, mk);
    v8bf lo, hi;
    #pragma unroll
    for (int cc = 0; cc < 16; cc++) {
      const float* xp = xb + (c0 + cc) * HWs;
      float v = tp.A00 * xp[tp.i00] + tp.A01 * xp[tp.i01] +
                tp.A10 * xp[tp.i10] + tp.A11 * xp[tp.i11];
      if (cc < 8) lo[cc] = f2bf(v); else hi[cc - 8] = f2bf(v);
    }
    bf16_t* dst = Bl + p * STR + k * 64 + c0;
    *(v8bf*)(dst)     = lo;
    *(v8bf*)(dst + 8) = hi;
  }
  __syncthreads();
  deform_mma_and_store(Al, Bl, b_dcn, out, b, y0, x0, t);
}

// ---------------------------------------------------------------------------
// Kernel 3a/3b: instance norm stats (deterministic two-stage) + apply/ReLU.
// ---------------------------------------------------------------------------
extern "C" __global__ __launch_bounds__(256)
void stats_part_kernel(const float* __restrict__ y, float2* __restrict__ part) {
  __shared__ float rs[256], rq[256];
  const int bo = blockIdx.y, ch = blockIdx.x, t = threadIdx.x;
  const float* p = y + (size_t)bo * HWs + ch * 2048;
  float s = 0.f, q = 0.f;
  for (int i = t; i < 2048; i += 256) { float v = p[i]; s += v; q += v * v; }
  rs[t] = s; rq[t] = q; __syncthreads();
  for (int off = 128; off > 0; off >>= 1) {
    if (t < off) { rs[t] += rs[t + off]; rq[t] += rq[t + off]; }
    __syncthreads();
  }
  if (t == 0) part[bo * 8 + ch] = make_float2(rs[0], rq[0]);
}

extern "C" __global__ __launch_bounds__(256)
void norm_apply_kernel(float* __restrict__ y, const float2* __restrict__ part) {
  const int bo = blockIdx.y, ch = blockIdx.x;
  float s = 0.f, q = 0.f;
  #pragma unroll
  for (int i = 0; i < 8; i++) { float2 v = part[bo * 8 + i]; s += v.x; q += v.y; }
  float mu  = s * (1.f / HWs);
  float var = q * (1.f / HWs) - mu * mu;
  float r   = rsqrtf(var + 1e-5f);
  float* p = y + (size_t)bo * HWs + ch * 2048;
  for (int i = threadIdx.x; i < 2048; i += 256) {
    float v = (p[i] - mu) * r;
    p[i] = v > 0.f ? v : 0.f;
  }
}

// ---------------------------------------------------------------------------
// Launcher. Inputs: x, w_off, b_off, w_dcn, b_dcn (all fp32).
// ---------------------------------------------------------------------------
extern "C" void kernel_launch(void* const* d_in, const int* in_sizes, int n_in,
                              void* d_out, int out_size, void* d_ws, size_t ws_size,
                              hipStream_t stream) {
  (void)in_sizes; (void)n_in; (void)out_size;
  const float* x     = (const float*)d_in[0];
  const float* w_off = (const float*)d_in[1];
  const float* b_off = (const float*)d_in[2];
  const float* w_dcn = (const float*)d_in[3];
  const float* b_dcn = (const float*)d_in[4];
  float* out = (float*)d_out;

  char* ws = (char*)d_ws;
  float*  offs  = (float*)ws;                         //  8*18*16384 f32 = 9.4 MB
  float*  maskp = (float*)(ws + 9437184);             //  8* 9*16384 f32 = 4.7 MB
  bf16_t* wboff = (bf16_t*)(ws + 14155776);           // 32*576 bf16
  bf16_t* wbdcn = (bf16_t*)(ws + 14192640);           // 64*576 bf16
  float2* part  = (float2*)(ws + 14266368);           // 512*8 float2 = 32 KB
  float*  xT    = (float*)(ws + 14299136);            // NHWC copy, 32 MB
  const size_t ws_need_nhwc = 14299136 + (size_t)8 * HWs * CIN * sizeof(float);

  prep_kernel<<<144, 256, 0, stream>>>(w_off, w_dcn, wboff, wbdcn);

  dim3 grid(WW / 16, HH / 4, 8);
  size_t sh1 = (size_t)(32 + 64) * STR * sizeof(bf16_t);   // ~112 KB
  offset_conv_kernel<<<grid, 256, sh1, stream>>>(x, wboff, b_off, offs, maskp);

  size_t sh2 = (size_t)(64 + 64) * STR * sizeof(bf16_t);   // ~146 KB
  if (ws_size >= ws_need_nhwc) {
    transpose_kernel<<<dim3(HWs / 64, 8), 256, 0, stream>>>(x, xT);
    deform_conv_nhwc_kernel<<<grid, 256, sh2, stream>>>(xT, wbdcn, b_dcn,
                                                        offs, maskp, out);
  } else {
    deform_conv_kernel<<<grid, 256, sh2, stream>>>(x, wbdcn, b_dcn,
                                                   offs, maskp, out);
  }

  stats_part_kernel<<<dim3(8, 8 * COUT), 256, 0, stream>>>(out, part);
  norm_apply_kernel<<<dim3(8, 8 * COUT), 256, 0, stream>>>(out, part);
}
